// RwkvSelfAttention_77515569758512
// MI455X (gfx1250) — compile-verified, matrix operand order checked by reading, not run on hardware
//
#include <hip/hip_runtime.h>

// ---------------------------------------------------------------------------
// RWKV self-attention, MI455X (gfx1250), wave32 + v_wmma_f32_16x16x32_bf16
// B=8, T=1024, C=2048
// ---------------------------------------------------------------------------

#define RB 8
#define RT 1024
#define RC 2048
#define RBT (RB * RT)          // 8192 rows of activation

typedef __attribute__((ext_vector_type(16))) __bf16 v16bf;
typedef __attribute__((ext_vector_type(8)))  __bf16 v8bf;
typedef __attribute__((ext_vector_type(8)))  float  v8f;
typedef __attribute__((ext_vector_type(4)))  int    v4i;

// Prefer the CDNA5 async global->LDS copy path (ASYNCcnt) when the toolchain
// exposes it; otherwise fall back to batched global_load_b128 + ds_store_b128.
#if defined(__has_builtin)
#if __has_builtin(__builtin_amdgcn_global_load_async_to_lds_b128) && \
    __has_builtin(__builtin_amdgcn_s_wait_asynccnt)
#define USE_ASYNC_LDS 1
#endif
#endif

#if defined(USE_ASYNC_LDS)
typedef __attribute__((address_space(1))) v4i* gptr_v4i;
typedef __attribute__((address_space(3))) v4i* lptr_v4i;
__device__ __forceinline__ void async_copy16(const __bf16* g, __bf16* l) {
    __builtin_amdgcn_global_load_async_to_lds_b128(
        (gptr_v4i)(g), (lptr_v4i)(l), 0, 0);
}
#endif

// Load a 16x32 bf16 fragment (A or B operand) for one lane.
// Lane layout per ISA 7.12.2: halves K..K+7 then K+16..K+23.
__device__ __forceinline__ v16bf ld_frag(const __bf16* p) {
    v8bf lo = *(const v8bf*)(p);
    v8bf hi = *(const v8bf*)(p + 16);
    return __builtin_shufflevector(lo, hi, 0,1,2,3,4,5,6,7,8,9,10,11,12,13,14,15);
}

// ---------------------------------------------------------------------------
// fp32 -> bf16 convert (weights)
// ---------------------------------------------------------------------------
__global__ __launch_bounds__(256) void f32_to_bf16_kernel(
        const float* __restrict__ in, __bf16* __restrict__ out, int n) {
    int i = blockIdx.x * 256 + threadIdx.x;
    if (i < n) out[i] = (__bf16)in[i];
}

// ---------------------------------------------------------------------------
// time-shift + time-mix:  out = x*mix + shift(x)*(1-mix)   (bf16 out)
// ---------------------------------------------------------------------------
__global__ __launch_bounds__(256) void time_mix_kernel(
        const float* __restrict__ x, const float* __restrict__ mix,
        __bf16* __restrict__ out) {
    int i = blockIdx.x * 256 + threadIdx.x;          // over B*T*C
    int c = i & (RC - 1);
    int t = (i >> 11) & (RT - 1);                    // C = 2^11
    float cur  = x[i];
    float prev = (t == 0) ? 0.0f : x[i - RC];
    float m    = mix[c];
    out[i] = (__bf16)(cur * m + prev * (1.0f - m));
}

// ---------------------------------------------------------------------------
// bf16 WMMA GEMM:  Out[M,N] = A[M,K] * W[N,K]^T
//   Block tile 256x128, K-step 64, 256 threads = 8 waves (4x2),
//   per-wave 64x64 => 4x4 accumulators of 16x16 (1:1 wmma:ds_load ratio).
//   Double-buffered LDS (120KB of the 320KB WGP pool); next tile staged with
//   async global->LDS copies while the current tile feeds 32 v_wmma.
// OUT_MODE: 0 = f32 store, 1 = bf16 store, 2 = sigmoid + bf16 store
// ---------------------------------------------------------------------------
template <int OUT_MODE>
__global__ __launch_bounds__(256) void gemm_bf16_wmma(
        const __bf16* __restrict__ A, const __bf16* __restrict__ W,
        void* __restrict__ Out, int M, int N, int K) {
    constexpr int BM    = 256;
    constexpr int BN    = 128;
    constexpr int BK    = 64;
    constexpr int LDSTR = BK + 16;                   // 80 elems = 160B rows, 16B aligned
    __shared__ __bf16 sA[2][BM * LDSTR];             // 2 x 40KB
    __shared__ __bf16 sB[2][BN * LDSTR];             // 2 x 20KB

    const int tid  = threadIdx.x;
    const int lane = tid & 31;
    const int wave = tid >> 5;
    const int wm   = (wave & 3) * 64;                // wave M offset in block
    const int wn   = (wave >> 2) * 64;               // wave N offset in block
    const int fm   = lane & 15;                      // fragment row/col within tile
    const int fh   = (lane >> 4) * 8;                // K half-select per ISA layout

    const int bm = blockIdx.y * BM;
    const int bn = blockIdx.x * BN;

    v8f acc[4][4];
#pragma unroll
    for (int mi = 0; mi < 4; ++mi)
#pragma unroll
        for (int ni = 0; ni < 4; ++ni)
            acc[mi][ni] = (v8f)(0.0f);

    // staging: A tile 256x64 = 2048 16B-chunks (8/thread), B tile 128x64 (4/thread)
#if defined(USE_ASYNC_LDS)
#define STAGE_ASYNC(bufi, kk0)                                                  \
    {                                                                           \
        _Pragma("unroll")                                                       \
        for (int i = 0; i < 8; ++i) {                                           \
            int lin = tid + i * 256;                                            \
            int row = lin >> 3;                                                 \
            int col = (lin & 7) * 8;                                            \
            async_copy16(A + (size_t)(bm + row) * K + (kk0) + col,              \
                         &sA[bufi][row * LDSTR + col]);                         \
        }                                                                       \
        _Pragma("unroll")                                                       \
        for (int i = 0; i < 4; ++i) {                                           \
            int lin = tid + i * 256;                                            \
            int row = lin >> 3;                                                 \
            int col = (lin & 7) * 8;                                            \
            async_copy16(W + (size_t)(bn + row) * K + (kk0) + col,              \
                         &sB[bufi][row * LDSTR + col]);                         \
        }                                                                       \
    }
#else
    uint4 ra[8], rw[4];
#define GLOAD(kk0)                                                              \
    {                                                                           \
        _Pragma("unroll")                                                       \
        for (int i = 0; i < 8; ++i) {                                           \
            int lin = tid + i * 256;                                            \
            int row = lin >> 3;                                                 \
            int col = (lin & 7) * 8;                                            \
            ra[i] = *(const uint4*)(A + (size_t)(bm + row) * K + (kk0) + col);  \
        }                                                                       \
        _Pragma("unroll")                                                       \
        for (int i = 0; i < 4; ++i) {                                           \
            int lin = tid + i * 256;                                            \
            int row = lin >> 3;                                                 \
            int col = (lin & 7) * 8;                                            \
            rw[i] = *(const uint4*)(W + (size_t)(bn + row) * K + (kk0) + col);  \
        }                                                                       \
    }
#define LSTORE(bufi)                                                            \
    {                                                                           \
        _Pragma("unroll")                                                       \
        for (int i = 0; i < 8; ++i) {                                           \
            int lin = tid + i * 256;                                            \
            int row = lin >> 3;                                                 \
            int col = (lin & 7) * 8;                                            \
            *(uint4*)(&sA[bufi][row * LDSTR + col]) = ra[i];                    \
        }                                                                       \
        _Pragma("unroll")                                                       \
        for (int i = 0; i < 4; ++i) {                                           \
            int lin = tid + i * 256;                                            \
            int row = lin >> 3;                                                 \
            int col = (lin & 7) * 8;                                            \
            *(uint4*)(&sB[bufi][row * LDSTR + col]) = rw[i];                    \
        }                                                                       \
    }
#endif

    int cur = 0;
#if defined(USE_ASYNC_LDS)
    STAGE_ASYNC(0, 0);
#else
    GLOAD(0);
    LSTORE(0);
#endif

    for (int k0 = 0; k0 < K; k0 += BK) {
        const bool has_next = (k0 + BK) < K;

#if defined(USE_ASYNC_LDS)
        if (has_next) {
            STAGE_ASYNC(cur ^ 1, k0 + BK);           // next tile in flight
            __builtin_amdgcn_s_wait_asynccnt(12);    // current tile resident
        } else {
            __builtin_amdgcn_s_wait_asynccnt(0);
        }
#else
        if (has_next) GLOAD(k0 + BK);                // global loads in flight
#endif
        __syncthreads();

        // --- compute: 2 k-steps of 32, 16 WMMA each ---
#pragma unroll
        for (int kk = 0; kk < BK; kk += 32) {
            v16bf afr[4], bfr[4];
#pragma unroll
            for (int mi = 0; mi < 4; ++mi)
                afr[mi] = ld_frag(&sA[cur][(wm + mi * 16 + fm) * LDSTR + kk + fh]);
#pragma unroll
            for (int ni = 0; ni < 4; ++ni)
                bfr[ni] = ld_frag(&sB[cur][(wn + ni * 16 + fm) * LDSTR + kk + fh]);
#pragma unroll
            for (int mi = 0; mi < 4; ++mi)
#pragma unroll
                for (int ni = 0; ni < 4; ++ni)
                    acc[mi][ni] = __builtin_amdgcn_wmma_f32_16x16x32_bf16(
                        false, afr[mi], false, bfr[ni],
                        (short)0, acc[mi][ni], false, false);
        }
        __syncthreads();

#if !defined(USE_ASYNC_LDS)
        if (has_next) LSTORE(cur ^ 1);
#endif
        cur ^= 1;
    }

    // --- epilogue: C/D layout -> lane n = lane&15, row = r + 8*(lane>>4) ---
    const int h8 = (lane >> 4) * 8;
    const int nc = lane & 15;
#pragma unroll
    for (int mi = 0; mi < 4; ++mi) {
#pragma unroll
        for (int ni = 0; ni < 4; ++ni) {
#pragma unroll
            for (int r = 0; r < 8; ++r) {
                int gm = bm + wm + mi * 16 + r + h8;
                int gn = bn + wn + ni * 16 + nc;
                float vv = acc[mi][ni][r];
                if (OUT_MODE == 0) {
                    ((float*)Out)[(size_t)gm * N + gn] = vv;
                } else {
                    if (OUT_MODE == 2) vv = 1.0f / (1.0f + __expf(-vv));
                    ((__bf16*)Out)[(size_t)gm * N + gn] = (__bf16)vv;
                }
            }
        }
    }
}

// ---------------------------------------------------------------------------
// WKV sequential scan with online max; one thread per (b,c) channel.
// rwkv = sigmoid(r) * wkv, written as bf16 for the final GEMM.
// ---------------------------------------------------------------------------
__global__ __launch_bounds__(256) void wkv_scan_kernel(
        const __bf16* __restrict__ k, const __bf16* __restrict__ v,
        const __bf16* __restrict__ sr,
        const float* __restrict__ time_decay, const float* __restrict__ time_first,
        __bf16* __restrict__ rwkv) {
    int idx = blockIdx.x * 256 + threadIdx.x;        // over B*C
    int c = idx & (RC - 1);
    float w = -__expf(time_decay[c]);
    float u = time_first[c];

    float num = 0.0f, den = 0.0f, mx = -1e38f;
    size_t base = (size_t)(idx >> 11) * RT * RC + c; // b*T*C + c
    for (int t = 0; t < RT; ++t) {
        float kt = (float)k[base];
        float vt = (float)v[base];
        float mo = fmaxf(mx, kt + u);
        float e1 = __expf(mx - mo);
        float e2 = __expf(kt + u - mo);
        float out = (e1 * num + e2 * vt) / (e1 * den + e2);
        float ms  = fmaxf(mx + w, kt);
        float e1s = __expf(mx + w - ms);
        float e2s = __expf(kt - ms);
        num = e1s * num + e2s * vt;
        den = e1s * den + e2s;
        mx  = ms;
        rwkv[base] = (__bf16)((float)sr[base] * out);
        base += RC;
    }
}

// ---------------------------------------------------------------------------
// driver
// ---------------------------------------------------------------------------
extern "C" void kernel_launch(void* const* d_in, const int* in_sizes, int n_in,
                              void* d_out, int out_size, void* d_ws, size_t ws_size,
                              hipStream_t stream) {
    (void)in_sizes; (void)n_in; (void)out_size; (void)ws_size;

    const float* x          = (const float*)d_in[0];
    const float* time_decay = (const float*)d_in[1];
    const float* time_first = (const float*)d_in[2];
    const float* mix_k      = (const float*)d_in[3];
    const float* mix_v      = (const float*)d_in[4];
    const float* mix_r      = (const float*)d_in[5];
    const float* Wk         = (const float*)d_in[6];
    const float* Wv         = (const float*)d_in[7];
    const float* Wr         = (const float*)d_in[8];
    const float* Wo         = (const float*)d_in[9];

    // workspace layout (bf16 buffers)
    const size_t ACT = (size_t)RBT * RC;             // 16.78M elems
    const size_t WEL = (size_t)RC * RC;              // 4.19M elems
    char* ws = (char*)d_ws;
    __bf16* Abuf = (__bf16*)(ws);                    // mix input / rwkv (reused)
    __bf16* kbuf = (__bf16*)(ws + 2 * ACT);
    __bf16* vbuf = (__bf16*)(ws + 4 * ACT);
    __bf16* sbuf = (__bf16*)(ws + 6 * ACT);          // sigmoid(r)
    __bf16* Wkb  = (__bf16*)(ws + 8 * ACT);
    __bf16* Wvb  = Wkb + WEL;
    __bf16* Wrb  = Wvb + WEL;
    __bf16* Wob  = Wrb + WEL;

    // 1) weights -> bf16
    {
        int n = (int)WEL;
        dim3 g((n + 255) / 256);
        f32_to_bf16_kernel<<<g, 256, 0, stream>>>(Wk, Wkb, n);
        f32_to_bf16_kernel<<<g, 256, 0, stream>>>(Wv, Wvb, n);
        f32_to_bf16_kernel<<<g, 256, 0, stream>>>(Wr, Wrb, n);
        f32_to_bf16_kernel<<<g, 256, 0, stream>>>(Wo, Wob, n);
    }

    dim3 mixGrid((unsigned)(ACT / 256));
    dim3 gemmGrid(RC / 128, RBT / 256);              // (N/BN, M/BM)

    // 2) k = mix_k(x) @ Wk^T
    time_mix_kernel<<<mixGrid, 256, 0, stream>>>(x, mix_k, Abuf);
    gemm_bf16_wmma<1><<<gemmGrid, 256, 0, stream>>>(Abuf, Wkb, kbuf, RBT, RC, RC);

    // 3) v = mix_v(x) @ Wv^T
    time_mix_kernel<<<mixGrid, 256, 0, stream>>>(x, mix_v, Abuf);
    gemm_bf16_wmma<1><<<gemmGrid, 256, 0, stream>>>(Abuf, Wvb, vbuf, RBT, RC, RC);

    // 4) sr = sigmoid(mix_r(x) @ Wr^T)
    time_mix_kernel<<<mixGrid, 256, 0, stream>>>(x, mix_r, Abuf);
    gemm_bf16_wmma<2><<<gemmGrid, 256, 0, stream>>>(Abuf, Wrb, sbuf, RBT, RC, RC);

    // 5) rwkv = sr * WKV-scan(k, v)   (Abuf reused as output)
    wkv_scan_kernel<<<dim3((RB * RC) / 256), 256, 0, stream>>>(
        kbuf, vbuf, sbuf, time_decay, time_first, Abuf);

    // 6) out = rwkv @ Wo^T  (fp32 into d_out)
    gemm_bf16_wmma<0><<<gemmGrid, 256, 0, stream>>>(Abuf, Wob, d_out, RBT, RC, RC);
}
